// LatentMoETransformerLayer_15659450761636
// MI455X (gfx1250) — matrix operand order repaired and checked by reading, not compiled, not run on hardware
//
#include <hip/hip_runtime.h>

// ---------------------------------------------------------------------------
// Types / constants
// ---------------------------------------------------------------------------
typedef __bf16 bf16_t;
typedef __attribute__((ext_vector_type(16))) __bf16 v16bf;
typedef __attribute__((ext_vector_type(8)))  __bf16 v8bf;
typedef __attribute__((ext_vector_type(8)))  float  v8f;

constexpr int CB  = 2;              // batch
constexpr int CS  = 1024;           // seq
constexpr int CD  = 768;            // d_model
constexpr int CH  = 12;             // heads
constexpr int CE  = 8;              // experts
constexpr int CF  = 3072;           // d_ff
constexpr int CHD = CD / CH;        // 64 head dim
constexpr int CNT = CB * CS;        // 2048 tokens
constexpr int C3D = 3 * CD;         // 2304

#define BM 256
#define BN 128
#define BK 32
#define LDTS (BK + 8)               // LDS tile row stride: 40 elems = 80 B (16B aligned)

__device__ __forceinline__ float gelu_f(float x) {
  const float c = 0.7978845608028654f;
  float t = tanhf(c * (x + 0.044715f * x * x * x));
  return 0.5f * x * (1.0f + t);
}

// ---------------------------------------------------------------------------
// Elementwise converters
// ---------------------------------------------------------------------------
__global__ void k_convert(const float* __restrict__ src, bf16_t* __restrict__ dst, long n) {
  long i = (long)blockIdx.x * blockDim.x + threadIdx.x;
  if (i < n) dst[i] = (bf16_t)src[i];
}

// Batched transpose+convert: src [E][R][C] f32 -> dst [E][C][R] bf16
__global__ void k_transpose_b(const float* __restrict__ src, bf16_t* __restrict__ dst,
                              int R, int C, long total) {
  long i = (long)blockIdx.x * blockDim.x + threadIdx.x;
  if (i >= total) return;
  const long per = (long)R * C;
  const long e = i / per;
  const long rem = i % per;
  const int r = (int)(rem / C);
  const int c = (int)(rem % C);
  dst[e * per + (long)c * R + r] = (bf16_t)src[i];
}

// ---------------------------------------------------------------------------
// LayerNorm (one 256-thread block per row of length CD)
// ---------------------------------------------------------------------------
__global__ __launch_bounds__(256) void k_layernorm(
    const float* __restrict__ x, const float* __restrict__ w, const float* __restrict__ b,
    bf16_t* __restrict__ out_bf, float* __restrict__ out_f, int D_) {
  const int row = blockIdx.x;
  const int tid = threadIdx.x;
  const float* xr = x + (long)row * D_;
  __shared__ float red[256];

  float s = 0.f;
  for (int i = tid; i < D_; i += 256) s += xr[i];
  red[tid] = s; __syncthreads();
  for (int o = 128; o > 0; o >>= 1) { if (tid < o) red[tid] += red[tid + o]; __syncthreads(); }
  const float mean = red[0] / (float)D_;
  __syncthreads();

  float v = 0.f;
  for (int i = tid; i < D_; i += 256) { float d = xr[i] - mean; v += d * d; }
  red[tid] = v; __syncthreads();
  for (int o = 128; o > 0; o >>= 1) { if (tid < o) red[tid] += red[tid + o]; __syncthreads(); }
  const float rstd = rsqrtf(red[0] / (float)D_ + 1e-5f);

  for (int i = tid; i < D_; i += 256) {
    float y = (xr[i] - mean) * rstd * w[i] + b[i];
    out_bf[(long)row * D_ + i] = (bf16_t)y;
    if (out_f) out_f[(long)row * D_ + i] = y;
  }
}

// ---------------------------------------------------------------------------
// Generic batched bf16 WMMA GEMM:  C[z] (+)= op(alpha * A[z]*B[z] + bias)
//
// B supplied TRANSPOSED (row-major [N][K], "BT") so both LDS tiles share the
// [rows][BK] geometry and every WMMA fragment is two aligned 16-byte chunks
// -> ds_load_b128 only.
//
// 256x128x32 block tile, 256 threads = 8 wave32 waves in a 4x2 grid; each
// wave computes a 64x64 patch = 4x4 v_wmma_f32_16x16x32_bf16 tiles per K
// step (16 WMMA : 16 ds_load_b128 -> 1 LDS fragment load per WMMA).
// Output offset: cBase = (z/batchMod)*sC0 + (z%batchMod)*sC1   (batchMod>=1)
// ---------------------------------------------------------------------------
__global__ __launch_bounds__(256) void k_gemm_bf16(
    const bf16_t* __restrict__ A, int lda, long sA,
    const bf16_t* __restrict__ BT, int ldb, long sB,
    float* __restrict__ C, int ldc, long sC0, long sC1, int batchMod,
    bf16_t* __restrict__ Cb,
    const float* __restrict__ bias,
    const float* __restrict__ residual,
    const float* __restrict__ rowScale, int rsStride,
    int M, int N, int K, float alpha, int actGelu, int accumulate) {
  __shared__ bf16_t As[BM * LDTS];
  __shared__ bf16_t Bs[BN * LDTS];

  const int z = blockIdx.z;
  const long aBase = (long)z * sA;
  const long bBase = (long)z * sB;
  const long cBase = (long)(z / batchMod) * sC0 + (long)(z % batchMod) * sC1;

  const int mBase = blockIdx.y * BM;
  const int nBase = blockIdx.x * BN;

  const int tid  = threadIdx.x;
  const int lane = tid & 31;
  const int wave = tid >> 5;
  const int wm   = wave >> 1;     // 0..3  (M direction, 64 rows each)
  const int wn   = wave & 1;      // 0..1  (N direction, 64 cols each)
  const int l16  = lane & 15;
  const int lh   = lane >> 4;

  v8f acc[4][4];
  const v8f vzero = {0.f, 0.f, 0.f, 0.f, 0.f, 0.f, 0.f, 0.f};
#pragma unroll
  for (int i = 0; i < 4; ++i)
#pragma unroll
    for (int j = 0; j < 4; ++j) acc[i][j] = vzero;

  // staging: thread tid copies A row tid (16 dwords); B row tid>>1 (8 dwords)
  const int bRow = tid >> 1;           // 0..127
  const int bCol = (tid & 1) * 8;      // u32 column offset

  int gmA = mBase + tid;  if (gmA >= M) gmA = M - 1;
  int gnB = nBase + bRow; if (gnB >= N) gnB = N - 1;

  const int nK = K / BK;               // K is a multiple of 32 in all uses
  for (int kb = 0; kb < nK; ++kb) {
    const int k0 = kb * BK;
    {
      const unsigned int* srcA =
          reinterpret_cast<const unsigned int*>(A + aBase + (long)gmA * lda + k0);
      unsigned int* dstA = reinterpret_cast<unsigned int*>(&As[tid * LDTS]);
#pragma unroll
      for (int i = 0; i < 16; ++i) dstA[i] = srcA[i];
      const unsigned int* srcB =
          reinterpret_cast<const unsigned int*>(BT + bBase + (long)gnB * ldb + k0);
      unsigned int* dstB = reinterpret_cast<unsigned int*>(&Bs[bRow * LDTS]);
#pragma unroll
      for (int i = 0; i < 8; ++i) dstB[bCol + i] = srcB[bCol + i];
    }
    if (kb + 1 < nK) {  // prefetch next K tile (global_prefetch_b8)
      __builtin_prefetch(A + aBase + (long)gmA * lda + k0 + BK, 0, 3);
      __builtin_prefetch(BT + bBase + (long)gnB * ldb + k0 + BK, 0, 3);
    }
    __syncthreads();

    // A fragments for this wave's 4 M tiles (each: K = lh*8+0..7, 16+lh*8+0..7)
    v16bf afr[4];
#pragma unroll
    for (int tm = 0; tm < 4; ++tm) {
      const bf16_t* ap = &As[(wm * 64 + tm * 16 + l16) * LDTS];
      const v8bf a0 = *reinterpret_cast<const v8bf*>(ap + lh * 8);
      const v8bf a1 = *reinterpret_cast<const v8bf*>(ap + 16 + lh * 8);
      afr[tm] = __builtin_shufflevector(a0, a1, 0, 1, 2, 3, 4, 5, 6, 7,
                                        8, 9, 10, 11, 12, 13, 14, 15);
    }
#pragma unroll
    for (int tn = 0; tn < 4; ++tn) {
      // B fragment (32x16): lane holds column n = l16, K = lh*8+0..7, 16+lh*8+0..7
      const bf16_t* bp = &Bs[(wn * 64 + tn * 16 + l16) * LDTS + lh * 8];
      const v8bf b0 = *reinterpret_cast<const v8bf*>(bp);
      const v8bf b1 = *reinterpret_cast<const v8bf*>(bp + 16);
      const v16bf bfr = __builtin_shufflevector(b0, b1, 0, 1, 2, 3, 4, 5, 6, 7,
                                                8, 9, 10, 11, 12, 13, 14, 15);
#pragma unroll
      for (int tm = 0; tm < 4; ++tm) {
        acc[tm][tn] = __builtin_amdgcn_wmma_f32_16x16x32_bf16(
            false, afr[tm], false, bfr, (short)0, acc[tm][tn], false, false);
      }
    }
    __syncthreads();
  }

  // ---- epilogue: C/D layout: VGPR r -> m = lh*8 + r, n = l16 ----
#pragma unroll
  for (int tm = 0; tm < 4; ++tm) {
#pragma unroll
    for (int tn = 0; tn < 4; ++tn) {
      const int gn = nBase + wn * 64 + tn * 16 + l16;
      if (gn >= N) continue;
#pragma unroll
      for (int r = 0; r < 8; ++r) {
        const int gm = mBase + wm * 64 + tm * 16 + lh * 8 + r;
        if (gm >= M) continue;
        float v = acc[tm][tn][r] * alpha;
        if (bias) v += bias[gn];
        if (actGelu) v = gelu_f(v);
        if (rowScale) v *= rowScale[(long)gm * rsStride];
        const long idx = cBase + (long)gm * ldc + gn;
        if (Cb) Cb[idx] = (bf16_t)v;
        if (C) {
          float o = v;
          if (residual) o += residual[idx];
          if (accumulate) o += C[idx];
          C[idx] = o;
        }
      }
    }
  }
}

// ---------------------------------------------------------------------------
// Split QKV f32 [N,3D] into bf16 Q[B,H,S,hd], K[B,H,S,hd] (=BT for Q@K^T),
// V^T[B,H,hd,S] (=BT for attn@V)
// ---------------------------------------------------------------------------
__global__ void k_split_qkv(const float* __restrict__ qkv,
                            bf16_t* __restrict__ Qb, bf16_t* __restrict__ Kb,
                            bf16_t* __restrict__ Vt) {
  long i = (long)blockIdx.x * blockDim.x + threadIdx.x;
  const long total = (long)CNT * CD;
  if (i >= total) return;
  const int d = (int)(i % CD);
  const long t = i / CD;
  const int s = (int)(t % CS);
  const int b = (int)(t / CS);
  const int h = d / CHD;
  const int dd = d % CHD;
  const long base = t * C3D;
  const long bh = (long)(b * CH + h);
  Qb[(bh * CS + s) * CHD + dd] = (bf16_t)qkv[base + d];
  Kb[(bh * CS + s) * CHD + dd] = (bf16_t)qkv[base + CD + d];
  Vt[(bh * CHD + dd) * CS + s] = (bf16_t)qkv[base + 2 * CD + d];
}

// ---------------------------------------------------------------------------
// Row softmax f32 -> bf16 (row length = len, one block per row)
// ---------------------------------------------------------------------------
__global__ __launch_bounds__(256) void k_softmax_rows(
    const float* __restrict__ scores, bf16_t* __restrict__ attn, int len) {
  const long row = blockIdx.x;
  const int tid = threadIdx.x;
  const float* r = scores + row * (long)len;
  bf16_t* o = attn + row * (long)len;
  __shared__ float red[256];

  float mx = -3.4e38f;
  for (int i = tid; i < len; i += 256) mx = fmaxf(mx, r[i]);
  red[tid] = mx; __syncthreads();
  for (int s = 128; s > 0; s >>= 1) { if (tid < s) red[tid] = fmaxf(red[tid], red[tid + s]); __syncthreads(); }
  mx = red[0]; __syncthreads();

  float sum = 0.f;
  for (int i = tid; i < len; i += 256) sum += __expf(r[i] - mx);
  red[tid] = sum; __syncthreads();
  for (int s = 128; s > 0; s >>= 1) { if (tid < s) red[tid] += red[tid + s]; __syncthreads(); }
  const float inv = 1.0f / red[0];

  for (int i = tid; i < len; i += 256) o[i] = (bf16_t)(__expf(r[i] - mx) * inv);
}

// ---------------------------------------------------------------------------
// Router: per-token logits/softmax/top-2 -> probs, sel, combine  (no atomics)
// ---------------------------------------------------------------------------
__global__ __launch_bounds__(256) void k_router(
    const float* __restrict__ h, const float* __restrict__ latent,
    const float* __restrict__ gw, const float* __restrict__ gb,
    float* __restrict__ probs, float* __restrict__ sel, float* __restrict__ comb) {
  const int t = blockIdx.x;
  const int b = t / CS;
  const int tid = threadIdx.x;
  float p[CE];
#pragma unroll
  for (int e = 0; e < CE; ++e) p[e] = 0.f;
  for (int d = tid; d < CD; d += 256) {
    const float xv = h[(long)t * CD + d] + latent[(long)b * CD + d];
#pragma unroll
    for (int e = 0; e < CE; ++e) p[e] += xv * gw[e * CD + d];
  }
  __shared__ float red[256];
  __shared__ float logits[CE];
  for (int e = 0; e < CE; ++e) {
    red[tid] = p[e]; __syncthreads();
    for (int s = 128; s > 0; s >>= 1) { if (tid < s) red[tid] += red[tid + s]; __syncthreads(); }
    if (tid == 0) logits[e] = red[0] + gb[e];
    __syncthreads();
  }
  if (tid == 0) {
    float mx = logits[0];
    for (int e = 1; e < CE; ++e) mx = fmaxf(mx, logits[e]);
    float pr[CE]; float sum = 0.f;
    for (int e = 0; e < CE; ++e) { pr[e] = __expf(logits[e] - mx); sum += pr[e]; }
    const float inv = 1.0f / sum;
    for (int e = 0; e < CE; ++e) pr[e] *= inv;
    int i0 = 0;
    for (int e = 1; e < CE; ++e) if (pr[e] > pr[i0]) i0 = e;
    int i1 = (i0 == 0) ? 1 : 0;
    for (int e = 0; e < CE; ++e) if (e != i0 && pr[e] > pr[i1]) i1 = e;
    const float ws = pr[i0] + pr[i1];
    for (int e = 0; e < CE; ++e) {
      probs[(long)t * CE + e] = pr[e];
      sel[(long)t * CE + e] = (e == i0 || e == i1) ? 1.f : 0.f;
      comb[(long)t * CE + e] = (e == i0) ? pr[i0] / ws : ((e == i1) ? pr[i1] / ws : 0.f);
    }
  }
}

// ---------------------------------------------------------------------------
// Balance loss: bal = E * sum_e (mean(sel)/K) * mean(probs)   (single block)
// ---------------------------------------------------------------------------
__global__ __launch_bounds__(256) void k_balance(
    const float* __restrict__ probs, const float* __restrict__ sel,
    float* __restrict__ outLoss, int Ntok) {
  const int tid = threadIdx.x;
  float aP[CE], aF[CE];
#pragma unroll
  for (int e = 0; e < CE; ++e) { aP[e] = 0.f; aF[e] = 0.f; }
  for (int i = tid; i < Ntok; i += 256)
#pragma unroll
    for (int e = 0; e < CE; ++e) {
      aP[e] += probs[(long)i * CE + e];
      aF[e] += sel[(long)i * CE + e];
    }
  __shared__ float red[256];
  __shared__ float sumP[CE], sumF[CE];
  for (int e = 0; e < CE; ++e) {
    red[tid] = aP[e]; __syncthreads();
    for (int s = 128; s > 0; s >>= 1) { if (tid < s) red[tid] += red[tid + s]; __syncthreads(); }
    if (tid == 0) sumP[e] = red[0];
    __syncthreads();
    red[tid] = aF[e]; __syncthreads();
    for (int s = 128; s > 0; s >>= 1) { if (tid < s) red[tid] += red[tid + s]; __syncthreads(); }
    if (tid == 0) sumF[e] = red[0];
    __syncthreads();
  }
  if (tid == 0) {
    float bal = 0.f;
    for (int e = 0; e < CE; ++e) {
      const float f = sumF[e] / ((float)Ntok * 2.0f);   // K = 2
      const float P = sumP[e] / (float)Ntok;
      bal += f * P;
    }
    outLoss[0] = (float)CE * bal;
  }
}

// ---------------------------------------------------------------------------
// Host orchestration
// ---------------------------------------------------------------------------
extern "C" void kernel_launch(void* const* d_in, const int* in_sizes, int n_in,
                              void* d_out, int out_size, void* d_ws, size_t ws_size,
                              hipStream_t stream) {
  (void)in_sizes; (void)n_in; (void)out_size; (void)ws_size;
  const float* x      = (const float*)d_in[0];
  const float* latent = (const float*)d_in[1];
  const float* ln1w   = (const float*)d_in[2];
  const float* ln1b   = (const float*)d_in[3];
  const float* qkvw   = (const float*)d_in[4];
  const float* qkvb   = (const float*)d_in[5];
  const float* outw   = (const float*)d_in[6];
  const float* outb   = (const float*)d_in[7];
  const float* ln2w   = (const float*)d_in[8];
  const float* ln2b   = (const float*)d_in[9];
  const float* gatew  = (const float*)d_in[10];
  const float* gateb  = (const float*)d_in[11];
  const float* w1     = (const float*)d_in[12];
  const float* b1     = (const float*)d_in[13];
  const float* w2     = (const float*)d_in[14];
  const float* b2     = (const float*)d_in[15];
  float* outp = (float*)d_out;

  char* ws = (char*)d_ws;
  size_t off = 0;
  auto take = [&](size_t bytes) -> char* {
    char* p = ws + off;
    off += (bytes + 255) & ~(size_t)255;
    return p;
  };
  // All B operands stored transposed [N][K] bf16.
  bf16_t* WqkvT = (bf16_t*)take((size_t)C3D * CD * 2);        // = qkv_w as-is
  bf16_t* WoutT = (bf16_t*)take((size_t)CD * CD * 2);         // = out_w as-is
  bf16_t* W1T   = (bf16_t*)take((size_t)CE * CF * CD * 2);    // [E][F][D]
  bf16_t* W2T   = (bf16_t*)take((size_t)CE * CD * CF * 2);    // [E][D][F]
  bf16_t* hB    = (bf16_t*)take((size_t)CNT * CD * 2);        // ln1 out bf16
  float*  hF    = (float*) take((size_t)CNT * CD * 4);        // ln2 out f32
  float*  qkvF  = (float*) take((size_t)CNT * C3D * 4);
  bf16_t* Qb    = (bf16_t*)take((size_t)CB * CH * CS * CHD * 2);
  bf16_t* Kb    = (bf16_t*)take((size_t)CB * CH * CS * CHD * 2);
  bf16_t* Vt    = (bf16_t*)take((size_t)CB * CH * CHD * CS * 2);
  float*  scF   = (float*) take((size_t)CB * CH * CS * CS * 4);
  bf16_t* atB   = (bf16_t*)take((size_t)CB * CH * CS * CS * 2);
  bf16_t* aoB   = (bf16_t*)take((size_t)CNT * CD * 2);
  bf16_t* tokB  = (bf16_t*)take((size_t)CNT * CD * 2);
  float*  probs = (float*) take((size_t)CNT * CE * 4);
  float*  selF  = (float*) take((size_t)CNT * CE * 4);
  float*  combF = (float*) take((size_t)CNT * CE * 4);
  bf16_t* h1B   = (bf16_t*)take((size_t)CNT * CF * 2);

  const dim3 blk(256);
  auto blocks1d = [](long n) { return dim3((unsigned)((n + 255) / 256)); };

  // --- weight conversion: qkv_w/out_w are already [N][K]; w1/w2 need transpose ---
  k_convert<<<blocks1d((long)C3D * CD), blk, 0, stream>>>(qkvw, WqkvT, (long)C3D * CD);
  k_convert<<<blocks1d((long)CD * CD), blk, 0, stream>>>(outw, WoutT, (long)CD * CD);
  k_transpose_b<<<blocks1d((long)CE * CD * CF), blk, 0, stream>>>(
      w1, W1T, CD, CF, (long)CE * CD * CF);   // [E][D][F] -> [E][F][D]
  k_transpose_b<<<blocks1d((long)CE * CF * CD), blk, 0, stream>>>(
      w2, W2T, CF, CD, (long)CE * CF * CD);   // [E][F][D] -> [E][D][F]

  // --- LN1 + QKV projection ---
  k_layernorm<<<dim3(CNT), blk, 0, stream>>>(x, ln1w, ln1b, hB, nullptr, CD);
  k_gemm_bf16<<<dim3(C3D / BN, CNT / BM, 1), blk, 0, stream>>>(
      hB, CD, 0, WqkvT, CD, 0, qkvF, C3D, 0, 0, 1, nullptr,
      qkvb, nullptr, nullptr, 0, CNT, C3D, CD, 1.0f, 0, 0);

  // --- attention ---
  k_split_qkv<<<blocks1d((long)CNT * CD), blk, 0, stream>>>(qkvF, Qb, Kb, Vt);
  // scores[b,h] = Q @ K^T / sqrt(hd)   (BT = K stored [S,hd])
  k_gemm_bf16<<<dim3(CS / BN, CS / BM, CB * CH), blk, 0, stream>>>(
      Qb, CHD, (long)CS * CHD, Kb, CHD, (long)CS * CHD, scF, CS, (long)CS * CS, 0, 1,
      nullptr, nullptr, nullptr, nullptr, 0, CS, CS, CHD, 0.125f, 0, 0);
  k_softmax_rows<<<dim3(CB * CH * CS), blk, 0, stream>>>(scF, atB, CS);
  // ao = attn @ V  (BT = V^T stored [hd,S]), written as [B,S,H*hd] bf16
  k_gemm_bf16<<<dim3(1, CS / BM, CB * CH), blk, 0, stream>>>(
      atB, CS, (long)CS * CS, Vt, CS, (long)CHD * CS, nullptr, CD, (long)CS * CD, CHD, CH,
      aoB, nullptr, nullptr, nullptr, 0, CS, CHD, CS, 1.0f, 0, 0);
  // x_attn = x + ao @ Wout^T + out_b  -> written into d_out
  k_gemm_bf16<<<dim3(CD / BN, CNT / BM, 1), blk, 0, stream>>>(
      aoB, CD, 0, WoutT, CD, 0, outp, CD, 0, 0, 1, nullptr,
      outb, x, nullptr, 0, CNT, CD, CD, 1.0f, 0, 0);

  // --- LN2 + router ---
  k_layernorm<<<dim3(CNT), blk, 0, stream>>>(outp, ln2w, ln2b, tokB, hF, CD);
  k_router<<<dim3(CNT), blk, 0, stream>>>(hF, latent, gatew, gateb, probs, selF, combF);

  // --- dense expert loop, accumulating into d_out (already holds x_attn) ---
  for (int e = 0; e < CE; ++e) {
    k_gemm_bf16<<<dim3(CF / BN, CNT / BM, 1), blk, 0, stream>>>(
        tokB, CD, 0, W1T + (long)e * CF * CD, CD, 0, nullptr, CF, 0, 0, 1, h1B,
        b1 + (long)e * CF, nullptr, nullptr, 0, CNT, CF, CD, 1.0f, /*gelu=*/1, 0);
    k_gemm_bf16<<<dim3(CD / BN, CNT / BM, 1), blk, 0, stream>>>(
        h1B, CF, 0, W2T + (long)e * CD * CF, CF, 0, outp, CD, 0, 0, 1, nullptr,
        b2 + (long)e * CD, nullptr, combF + e, CE, CNT, CD, CF, 1.0f, 0, /*accum=*/1);
  }

  // --- balance loss ---
  k_balance<<<dim3(1), blk, 0, stream>>>(probs, selF, outp + (long)CNT * CD, CNT);
}
